// ClassicalEncoderDecoder_57080115363917
// MI455X (gfx1250) — matrix operand order repaired and checked by reading, not compiled, not run on hardware
//
#include <hip/hip_runtime.h>
#include <hip/hip_bf16.h>
#include <math.h>

// ---------------------------------------------------------------------------
// ClassicalEncoderDecoder for MI455X (gfx1250).
//
// The ring-rotation chain is linear => fold the 4 encoder blocks into one
// 1024x1024 matrix M_enc (likewise M_dec), then two f16 WMMA GEMMs:
//   bottleneck = (1-w) * (x @ M_enc) + w * hidden    -> d_out[0 : B*N]
//   out        = bottleneck @ M_dec                   -> d_out[B*N : 2*B*N]
// B = 8192, N = K = 1024.  A stripes are staged into LDS via the Tensor Data
// Mover (TDM); B fragments are double-buffered in registers
// (__launch_bounds__(256,2) so the ~210 live VGPRs fit without spilling).
// ---------------------------------------------------------------------------

#define RN    1024          // ring dimension (n) == GEMM K == GEMM N
#define RB    8192          // batch rows (B)
#define NBLK  4             // ring blocks per matrix
#define MBLK  64            // M rows per workgroup (TDM tile height)
#define NBLKC 256           // N cols per workgroup

typedef _Float16 v4h  __attribute__((ext_vector_type(4)));
typedef _Float16 v8h  __attribute__((ext_vector_type(8)));
typedef _Float16 v16h __attribute__((ext_vector_type(16)));
typedef float    v8f  __attribute__((ext_vector_type(8)));
typedef unsigned int v4u __attribute__((ext_vector_type(4)));
typedef int      v8i  __attribute__((ext_vector_type(8)));
typedef int      v4i  __attribute__((ext_vector_type(4)));

// ---------------------------------------------------------------------------
// Kernel 1: cos/sin tables, cs[((mat*4+blk)*1024 + k)*2 + {0,1}] = {cos,sin}
// ---------------------------------------------------------------------------
__global__ void ring_cs(const float* __restrict__ ae,
                        const float* __restrict__ ad,
                        float* __restrict__ cs) {
    int t = blockIdx.x * blockDim.x + threadIdx.x;   // 0..8191
    float a = (t < NBLK * RN) ? ae[t] : ad[t - NBLK * RN];
    cs[2 * t + 0] = cosf(a);
    cs[2 * t + 1] = sinf(a);
}

// ---------------------------------------------------------------------------
// Kernel 2: build M_enc^T / M_dec^T by running the exact reference recurrence
// on identity rows.  T[m*RN + i] = M[i][m] (K-contiguous for the B-fragment).
// Thread i exclusively owns column i => no synchronization, in-place safe.
// ---------------------------------------------------------------------------
template <bool FIRST>
__device__ __forceinline__ void ring_block(_Float16* __restrict__ T,
                                           const float* __restrict__ c2,
                                           int i) {
    const int n = RN;
    float x0, xi;
    if (FIRST) { x0 = (i == 0)     ? 1.f : 0.f; }
    else       { x0 = (float)T[0 * n + i]; }
    float c = c2[2 * (n - 1) + 0], s = c2[2 * (n - 1) + 1];
    if (FIRST) { xi = (i == n - 1) ? 1.f : 0.f; }
    else       { xi = (float)T[(n - 1) * n + i]; }
    float x0val = s * xi + c * x0;          // new x[0], consumed by step k=0
    float carry = c * xi - s * x0;          // new x[n-1], read by step n-2
    for (int k = n - 2; k >= 1; --k) {
        if (FIRST) { xi = (k == i) ? 1.f : 0.f; }
        else       { xi = (float)T[k * n + i]; }
        c = c2[2 * k + 0]; s = c2[2 * k + 1];
        T[(k + 1) * n + i] = (_Float16)(s * xi + c * carry);  // final x[k+1]
        carry = c * xi - s * carry;                            // new x[k]
    }
    c = c2[0]; s = c2[1];
    T[1 * n + i] = (_Float16)(s * x0val + c * carry);
    T[0 * n + i] = (_Float16)(c * x0val - s * carry);
}

__global__ void ring_build(const float* __restrict__ cs,
                           _Float16* __restrict__ ThE,
                           _Float16* __restrict__ ThD) {
    int tid = blockIdx.x * blockDim.x + threadIdx.x;  // 0..2047
    int mat = tid >> 10;                              // 0 = enc, 1 = dec
    int i   = tid & (RN - 1);
    _Float16* T = mat ? ThD : ThE;
    const float* base = cs + (size_t)mat * NBLK * RN * 2;
    ring_block<true>(T, base + 0 * RN * 2, i);
    for (int b = 1; b < NBLK; ++b)
        ring_block<false>(T, base + (size_t)b * RN * 2, i);
}

// ---------------------------------------------------------------------------
// Kernel 3: x f32 -> f16 (float4 -> 4 halves)
// ---------------------------------------------------------------------------
__global__ void cvt_f32_f16(const float* __restrict__ in,
                            _Float16* __restrict__ out, int n4) {
    int t = blockIdx.x * blockDim.x + threadIdx.x;
    if (t < n4) {
        float4 v = ((const float4*)in)[t];
        v4h o = {(_Float16)v.x, (_Float16)v.y, (_Float16)v.z, (_Float16)v.w};
        ((v4h*)out)[t] = o;
    }
}

// ---------------------------------------------------------------------------
// WMMA GEMM helpers.  Fragment layouts per CDNA5 ISA 7.12.2 (wave32):
//   A 16x32 f16: lane L -> row (L&15); halves: K = kk + (L>=16?8:0) + h (h<8)
//                                      K = kk + 16 + (L>=16?8:0) + h-8 (h>=8)
//   B 32x16 f16: lane L -> col (L&15); K = kk + (L>=16?8:0) + h (+8 if h>=8), same as A
//   C/D 16x16 f32: elem r -> row r + (L>=16?8:0), col (L&15)
// ---------------------------------------------------------------------------
__device__ __forceinline__ v16h join16(v8h a, v8h b) {
    return __builtin_shufflevector(a, b, 0, 1, 2, 3, 4, 5, 6, 7,
                                   8, 9, 10, 11, 12, 13, 14, 15);
}

__device__ __forceinline__ v16h load_a_lds(const _Float16* __restrict__ sm,
                                           int rowBase, int kk, int hi) {
    const _Float16* ap = sm + rowBase + kk + hi * 8;
    return join16(*(const v8h*)ap, *(const v8h*)(ap + 16));
}

__device__ __forceinline__ void load_b8(const _Float16* __restrict__ Bt,
                                        int N0, int r16, int hi, int kk,
                                        v16h bf[8]) {
#pragma unroll
    for (int t = 0; t < 8; ++t) {
        const _Float16* bp = Bt + (size_t)(N0 + t * 16 + r16) * RN + kk + hi * 8;
        bf[t] = join16(*(const v8h*)bp, *(const v8h*)(bp + 16));
    }
}

__device__ __forceinline__ void wmma8(v16h af, const v16h bf[8], v8f acc[8]) {
#pragma unroll
    for (int t = 0; t < 8; ++t)
        acc[t] = __builtin_amdgcn_wmma_f32_16x16x32_f16(
            false, af, false, bf[t], (short)0, acc[t], false, false);
}

// ---------------------------------------------------------------------------
// Kernel 4/5: C[r][m] = sum_i A[r][i] * Bt[m][i].  Block = 256 thr = 8 waves
// = 4 M-subtiles x 2 N-groups => block tile 64M x 256N; wave tile 16M x 128N,
// K-loop of v_wmma_f32_16x16x32_f16 with register double-buffered B.
// The block's 64x1024 f16 A stripe (128 KB) is DMA'd into LDS once by the
// Tensor Data Mover (D# built per ISA 8.3/8.4), synced with s_wait_tensorcnt.
// __launch_bounds__(256, 2): 2 waves/SIMD -> enough VGPR budget, no spills.
// ---------------------------------------------------------------------------
template <bool BLEND>
__global__ void __launch_bounds__(256, 2)
ring_gemm(const _Float16* __restrict__ A,
          const _Float16* __restrict__ Bt,
          float* __restrict__ outF,
          _Float16* __restrict__ outH,
          const float* __restrict__ hidden,
          const float* __restrict__ hwp) {
    extern __shared__ _Float16 smem[];          // [MBLK][RN] f16 = 128 KB
    const int wave = threadIdx.x >> 5;
    const int lane = threadIdx.x & 31;
    const int r16  = lane & 15;
    const int hi   = lane >> 4;
    const int mSub = wave >> 1;                 // 0..3
    const int ng   = wave & 1;                  // 0..1
    const int Mblk = blockIdx.x * MBLK;
    const int N0   = blockIdx.y * NBLKC + ng * 128;

    // --- TDM: A[Mblk .. Mblk+63][0..1023] -> LDS (row-major, contiguous) ---
    if (threadIdx.x < 32) {
        unsigned long long ga =
            (unsigned long long)(const void*)(A + (size_t)Mblk * RN);
        unsigned int lds_off = (unsigned int)(unsigned long long)(void*)smem;
        // D# group0: count=1 | lds_addr | global_addr[56:0] | type=2
        v4u g0 = {1u, lds_off, (unsigned int)ga,
                  (unsigned int)(ga >> 32) | 0x80000000u};
        // D# group1: data_size=2B | tensor_dim0=1024 | tensor_dim1=64 |
        //            tile_dim0=1024 | tile_dim1=64 | dim0_stride=1024
        v8i g1 = {1 << 16, 1024 << 16, MBLK << 16, 1024 << 16,
                  MBLK, 1024, 0, 0};
        v4i z4 = {0, 0, 0, 0};
        v8i z8 = {0, 0, 0, 0, 0, 0, 0, 0};
        __builtin_amdgcn_tensor_load_to_lds(g0, g1, z4, z4, z8, 0);
        __builtin_amdgcn_s_wait_tensorcnt(0);
    }
    __syncthreads();

    const int rowBase = (mSub * 16 + r16) * RN;
    v8f acc[8] = {};
    v16h bA[8], bB[8];

    load_b8(Bt, N0, r16, hi, 0, bA);
    for (int kk = 0; kk < RN; kk += 64) {
        __builtin_prefetch(Bt + (size_t)(N0 + r16) * RN + kk + 64, 0, 3);
        load_b8(Bt, N0, r16, hi, kk + 32, bB);       // prefetch next chunk
        v16h af = load_a_lds(smem, rowBase, kk, hi);
        wmma8(af, bA, acc);
        if (kk + 64 < RN) load_b8(Bt, N0, r16, hi, kk + 64, bA);
        af = load_a_lds(smem, rowBase, kk + 32, hi);
        wmma8(af, bB, acc);
    }

    float w = 0.f;
    if (BLEND) { float h = hwp[0]; w = 1.f / (1.f + expf(-h)); }

#pragma unroll
    for (int t = 0; t < 8; ++t) {
        int col = N0 + t * 16 + r16;
        float hv = BLEND ? hidden[col] : 0.f;
#pragma unroll
        for (int r = 0; r < 8; ++r) {
            int row = Mblk + mSub * 16 + r + hi * 8;
            float v = acc[t][r];
            if (BLEND) v = (1.f - w) * v + w * hv;
            outF[(size_t)row * RN + col] = v;
            if (outH) outH[(size_t)row * RN + col] = (_Float16)v;
        }
    }
}

// ---------------------------------------------------------------------------
// Host launcher.  Inputs: x[8192*1024] f32, angles_enc[4*1024],
// angles_dec[4*1024], hidden_weight[1], hidden_state[1024].
// d_out: 2 * 8192 * 1024 f32 (bottleneck, out).
// ws (~36.1 MiB): cs table | ThE f16 | ThD f16 | Xh f16 | Bh f16
// ---------------------------------------------------------------------------
extern "C" void kernel_launch(void* const* d_in, const int* in_sizes, int n_in,
                              void* d_out, int out_size, void* d_ws,
                              size_t ws_size, hipStream_t stream) {
    const float* x  = (const float*)d_in[0];
    const float* ae = (const float*)d_in[1];
    const float* ad = (const float*)d_in[2];
    const float* hw = (const float*)d_in[3];
    const float* hs = (const float*)d_in[4];
    float* outF = (float*)d_out;

    char* ws = (char*)d_ws;
    float*    cs  = (float*)(ws);                       // 64 KiB
    _Float16* ThE = (_Float16*)(ws + 65536);            // 2 MiB
    _Float16* ThD = ThE + (size_t)RN * RN;              // 2 MiB
    _Float16* Xh  = ThD + (size_t)RN * RN;              // 16 MiB
    _Float16* Bh  = Xh + (size_t)RB * RN;               // 16 MiB
    (void)in_sizes; (void)n_in; (void)out_size; (void)ws_size;

    // 1) cos/sin tables (8192 angles)
    ring_cs<<<32, 256, 0, stream>>>(ae, ad, cs);
    // 2) fold ring blocks into M_enc^T / M_dec^T
    ring_build<<<8, 256, 0, stream>>>(cs, ThE, ThD);
    // 3) x -> f16
    cvt_f32_f16<<<(RB * RN / 4 + 255) / 256, 256, 0, stream>>>(
        x, Xh, RB * RN / 4);

    dim3 grid(RB / MBLK, RN / NBLKC);                   // 128 x 4
    size_t ldsBytes = (size_t)MBLK * RN * sizeof(_Float16);  // 128 KB
    // 4) bottleneck = (1-w)*(x @ M_enc) + w*h  (f32 out + f16 copy for GEMM2)
    ring_gemm<true><<<grid, 256, ldsBytes, stream>>>(Xh, ThE, outF, Bh, hs, hw);
    // 5) out = bottleneck @ M_dec
    ring_gemm<false><<<grid, 256, ldsBytes, stream>>>(
        Bh, ThD, outF + (size_t)RB * RN, (_Float16*)nullptr, hs, hw);
}
